// DecoderModule_3289944949534
// MI455X (gfx1250) — compile-verified, matrix operand order checked by reading, not run on hardware
//
#include <hip/hip_runtime.h>

// ---------------------------------------------------------------------------
// CDNA5 (gfx1250) LSTM decoder. All heavy GEMMs use v_wmma_f32_16x16x32_f16.
// Weights are converted to f16 once per launch (W_out / W_att pre-transposed
// so the WMMA "B" operand reads contiguous rows). fp32 accumulate everywhere.
// Each GEMM wave computes TWO adjacent 16x16 N-tiles sharing one A fetch:
// 6 x global_load_b128 + 2 x v_wmma per K=32 chunk.
// ---------------------------------------------------------------------------

typedef __attribute__((ext_vector_type(16))) _Float16 v16h;
typedef __attribute__((ext_vector_type(8)))  _Float16 v8h;
typedef __attribute__((ext_vector_type(8)))  float    v8f;

// Problem constants (match reference)
#define NB   64           // batch
#define NV   32000        // vocab
#define NE   512          // embed
#define NH   1024         // hidden
#define NU   16           // UNK slots
#define NT   30           // steps
#define KCAT 3072         // IN_DIM(2048) + H(1024)
#define N4H  4096         // 4*H

// -------------------------- workspace layout (bytes) -----------------------
static constexpr size_t OFF_WCAT   = 0;                                          // f16 [4096,3072]
static constexpr size_t OFF_WOUTT  = OFF_WCAT  + (size_t)N4H * KCAT * 2;         // f16 [32000,1024]
static constexpr size_t OFF_WATTT  = OFF_WOUTT + (size_t)NV  * NH   * 2;         // f16 [512,1024]
static constexpr size_t OFF_XCAT   = OFF_WATTT + (size_t)NE  * NH   * 2;         // f16 [64,3072]
static constexpr size_t OFF_Z      = OFF_XCAT  + (size_t)NB  * KCAT * 2;         // f32 [64,4096]
static constexpr size_t OFF_H      = OFF_Z     + (size_t)NB  * N4H  * 4;         // f32 [64,1024]
static constexpr size_t OFF_C      = OFF_H     + (size_t)NB  * NH   * 4;         // f32 [64,1024]
static constexpr size_t OFF_H16    = OFF_C     + (size_t)NB  * NH   * 4;         // f16 [64,1024]
static constexpr size_t OFF_Q      = OFF_H16   + (size_t)NB  * NH   * 2;         // f32 [64,512]
static constexpr size_t OFF_LOGITS = OFF_Q     + (size_t)NB  * NE   * 4;         // f32 [64,32000]
static constexpr size_t OFF_SCORES = OFF_LOGITS+ (size_t)NB  * NV   * 4;         // f32 [64,17]
static constexpr size_t OFF_UW     = OFF_SCORES+ (size_t)NB  * 17   * 4;         // f32 [64,17]
static constexpr size_t OFF_LUNK16 = OFF_UW    + (size_t)NB  * 17   * 4;         // f16 [64,512]
static constexpr size_t OFF_UMASK  = OFF_LUNK16+ (size_t)NB  * NE   * 2;         // f32 [64,16]
static constexpr size_t OFF_REACH  = OFF_UMASK + (size_t)NB  * NU   * 4;         // i32 [64]
static constexpr size_t OFF_LEN    = OFF_REACH + 256;                            // i32 [64]
static constexpr size_t OFF_LAST   = OFF_LEN   + 256;                            // i32 [64]
static constexpr size_t OFF_WARG   = OFF_LAST  + 256;                            // i32 [64]
static constexpr size_t OFF_UARG   = OFF_WARG  + 256;                            // i32 [64]

// d_out layout (flat float, return order)
static constexpr size_t OUT_UW_BASE   = (size_t)NB * NT * NV;        // after all_outputs
static constexpr size_t OUT_PRED_BASE = OUT_UW_BASE + (size_t)NB * NT * (NU + 1);
static constexpr size_t OUT_LEN_BASE  = OUT_PRED_BASE + (size_t)NB * NT;

// -------------------------- weight preparation -----------------------------

// Wcat[n][k] = k<2048 ? W_ih[n][k] : W_hh[n][k-2048]   (f32 -> f16)
__global__ void k_build_wcat(const float* __restrict__ Wih,
                             const float* __restrict__ Whh,
                             _Float16* __restrict__ Wcat) {
  size_t idx = (size_t)blockIdx.x * 256 + threadIdx.x;       // exactly 4096*3072
  int n = (int)(idx / KCAT);
  int k = (int)(idx % KCAT);
  float v = (k < 2048) ? Wih[(size_t)n * 2048 + k]
                       : Whh[(size_t)n * 1024 + (k - 2048)];
  Wcat[idx] = (_Float16)v;
}

// Wt[n][k] = W[k][n]  (W row-major [K,N] f32 -> Wt row-major [N,K] f16)
__global__ void k_transpose_f16(const float* __restrict__ W,
                                _Float16* __restrict__ Wt, int K, int N) {
  size_t idx = (size_t)blockIdx.x * 256 + threadIdx.x;
  if (idx >= (size_t)N * K) return;
  int n = (int)(idx / K);
  int k = (int)(idx % K);
  Wt[idx] = (_Float16)W[(size_t)k * N + n];
}

// -------------------------- state init -------------------------------------
__global__ void k_init(float* h, float* c, _Float16* h16, _Float16* lunk16,
                       float* umask, int* reached, int* lengths, int* last_out) {
  int t = blockIdx.x * 256 + threadIdx.x;   // 65536 threads
  if (t < NB * NH) { h[t] = 0.f; c[t] = 0.f; h16[t] = (_Float16)0.f; }
  if (t < NB * NE) lunk16[t] = (_Float16)0.f;
  if (t < NB * NU) umask[t] = 1.f;
  if (t < NB) { reached[t] = 0; lengths[t] = 0; last_out[t] = 1; /* SOS */ }
}

// -------------------------- per-step kernels -------------------------------

// x = [emb(last) | semantics | styles | last_unk | h]  (f16, K=3072)
__global__ void k_build_x(const float* __restrict__ emb_table,
                          const float* __restrict__ semantics,
                          const float* __restrict__ styles,
                          const _Float16* __restrict__ lunk16,
                          const _Float16* __restrict__ h16,
                          const int* __restrict__ last_out,
                          const int* __restrict__ reached,
                          int* __restrict__ lengths,
                          _Float16* __restrict__ xcat) {
  int b = blockIdx.x, tid = threadIdx.x;
  if (tid == 0) lengths[b] += (1 - reached[b]);               // lengths bookkeeping
  const float* emb = emb_table + (size_t)last_out[b] * NE;
  for (int k = tid; k < KCAT; k += 256) {
    _Float16 v;
    if      (k < 512)  v = (_Float16)emb[k];
    else if (k < 1024) v = (_Float16)semantics[b * NE + (k - 512)];
    else if (k < 1536) v = (_Float16)styles[b * NE + (k - 1024)];
    else if (k < 2048) v = lunk16[b * NE + (k - 1536)];
    else               v = h16[b * NH + (k - 2048)];
    xcat[(size_t)b * KCAT + k] = v;
  }
}

// C[64,N] = A[64,K]f16 * Wt[N,K]^T f16 (+bias), fp32 accumulate via WMMA.
// One wave per 16(M) x 32(N) macro-tile: two 16x16 WMMA tiles share one A
// fetch per K=32 chunk (6 x b128 loads + 2 x wmma). K%32==0, N%32==0, M=64.
// A-operand ISA layout (16-bit A 16x32): lanes 0-15 hold K=0-7,16-23;
// lanes 16-31 hold K=8-15,24-31. B-operand mirrors with N on lanes.
__global__ __launch_bounds__(32)
void k_gemm_wmma(const _Float16* __restrict__ A, const _Float16* __restrict__ Wt,
                 const float* __restrict__ bias, float* __restrict__ C,
                 int K, int N) {
  const int lane = threadIdx.x;
  const int n0 = blockIdx.x * 32;                   // two N tiles: n0, n0+16
  const int m0 = blockIdx.y * 16;
  const int l15  = lane & 15;
  const int koff = (lane >> 4) * 8;                 // 0 for lanes 0-15, 8 for 16-31
  const _Float16* arow  = A  + (size_t)(m0 + l15) * K + koff;
  const _Float16* brow0 = Wt + (size_t)(n0 + l15) * K + koff;
  const _Float16* brow1 = Wt + (size_t)(n0 + 16 + l15) * K + koff;
  v8f acc0 = {0.f, 0.f, 0.f, 0.f, 0.f, 0.f, 0.f, 0.f};
  v8f acc1 = {0.f, 0.f, 0.f, 0.f, 0.f, 0.f, 0.f, 0.f};
  for (int k0 = 0; k0 < K; k0 += 32) {
    __builtin_prefetch(brow0 + k0 + 128, 0, 1);     // global_prefetch_b8 on weights
    __builtin_prefetch(brow1 + k0 + 128, 0, 1);
    v8h a0 = *(const v8h*)(arow + k0);
    v8h a1 = *(const v8h*)(arow + k0 + 16);
    v8h p0 = *(const v8h*)(brow0 + k0);
    v8h p1 = *(const v8h*)(brow0 + k0 + 16);
    v8h q0 = *(const v8h*)(brow1 + k0);
    v8h q1 = *(const v8h*)(brow1 + k0 + 16);
    v16h av  = __builtin_shufflevector(a0, a1, 0,1,2,3,4,5,6,7,8,9,10,11,12,13,14,15);
    v16h bv0 = __builtin_shufflevector(p0, p1, 0,1,2,3,4,5,6,7,8,9,10,11,12,13,14,15);
    v16h bv1 = __builtin_shufflevector(q0, q1, 0,1,2,3,4,5,6,7,8,9,10,11,12,13,14,15);
    acc0 = __builtin_amdgcn_wmma_f32_16x16x32_f16(
        false, av, false, bv0, (short)0, acc0, false, false);
    acc1 = __builtin_amdgcn_wmma_f32_16x16x32_f16(
        false, av, false, bv1, (short)0, acc1, false, false);
  }
  // D layout: VGPR r -> M = r + 8*(lane>=16); N = lane&15
  const int mb = (lane >> 4) * 8;
  const float bv0 = bias ? bias[n0 + l15] : 0.f;
  const float bv1 = bias ? bias[n0 + 16 + l15] : 0.f;
  #pragma unroll
  for (int r = 0; r < 8; ++r) {
    C[(size_t)(m0 + mb + r) * N + n0 + l15]      = acc0[r] + bv0;
    C[(size_t)(m0 + mb + r) * N + n0 + 16 + l15] = acc1[r] + bv1;
  }
}

// LSTM pointwise: z[64,4096] -> h,c (fp32) + h16
__global__ void k_lstm(const float* __restrict__ z, float* __restrict__ h,
                       float* __restrict__ c, _Float16* __restrict__ h16) {
  int idx = blockIdx.x * 256 + threadIdx.x;     // 65536
  int b = idx >> 10, j = idx & 1023;
  const float* zb = z + (size_t)b * N4H;
  float zi = zb[j], zf = zb[1024 + j], zg = zb[2048 + j], zo = zb[3072 + j];
  float si = 1.f / (1.f + __expf(-zi));
  float sf = 1.f / (1.f + __expf(-zf));
  float so = 1.f / (1.f + __expf(-zo));
  float cn = sf * c[idx] + si * tanhf(zg);
  float hn = so * tanhf(cn);
  c[idx] = cn; h[idx] = hn; h16[idx] = (_Float16)hn;
}

// scores[b][u]: u==0 -> h.w_nounk ; u>=1 -> q . UNK_embeds[b][u-1]
__global__ __launch_bounds__(32)
void k_scores(const float* __restrict__ q, const float* __restrict__ h,
              const float* __restrict__ UNK_embeds,
              const float* __restrict__ w_nounk, float* __restrict__ scores) {
  int u = blockIdx.x, b = blockIdx.y, lane = threadIdx.x;
  float p = 0.f;
  if (u == 0) {
    const float* hb = h + (size_t)b * NH;
    for (int i = lane; i < NH; i += 32) p += hb[i] * w_nounk[i];
  } else {
    const float* qb = q + (size_t)b * NE;
    const float* eb = UNK_embeds + ((size_t)b * NU + (u - 1)) * NE;
    for (int i = lane; i < NE; i += 32) p += qb[i] * eb[i];
  }
  for (int off = 16; off > 0; off >>= 1) p += __shfl_down(p, off);
  if (lane == 0) scores[b * 17 + u] = p;
}

// softmax over [no_unk | masked scores], last_unk = sum uw[1:]*UNK_embeds
__global__ __launch_bounds__(32)
void k_unk_softmax(const float* __restrict__ scores,
                   const int* __restrict__ UNK_lengths,
                   const float* __restrict__ umask,
                   const float* __restrict__ UNK_embeds,
                   float* __restrict__ uw, _Float16* __restrict__ lunk16,
                   int* __restrict__ unk_argmax, float* __restrict__ d_uw) {
  int b = blockIdx.x, lane = threadIdx.x;
  __shared__ float s[17];
  if (lane < 17) {
    float v = scores[b * 17 + lane];
    if (lane >= 1) {
      bool valid = ((lane - 1) < UNK_lengths[b]) && (umask[b * NU + lane - 1] > 0.f);
      if (!valid) v = -1e9f;
    }
    s[lane] = v;
  }
  __syncthreads();
  if (lane == 0) {                       // serial, deterministic (17 elems)
    float mx = s[0];
    for (int i = 1; i < 17; ++i) mx = fmaxf(mx, s[i]);
    float e[17], sum = 0.f;
    for (int i = 0; i < 17; ++i) { e[i] = __expf(s[i] - mx); sum += e[i]; }
    float inv = 1.f / sum;
    for (int i = 0; i < 17; ++i) s[i] = e[i] * inv;
    int am = 0; float best = -1.f;
    for (int i = 1; i < 17; ++i) if (s[i] > best) { best = s[i]; am = i - 1; }
    unk_argmax[b] = am;
  }
  __syncthreads();
  if (lane < 17) {
    uw[b * 17 + lane] = s[lane];
    d_uw[(size_t)b * (NT * 17) + lane] = s[lane];     // all_UNK_weights[b][t][:]
  }
  const float* eb = UNK_embeds + (size_t)b * NU * NE;
  for (int e2 = lane; e2 < NE; e2 += 32) {
    float acc = 0.f;
    for (int u = 0; u < NU; ++u) acc += s[u + 1] * eb[(size_t)u * NE + e2];
    lunk16[b * NE + e2] = (_Float16)acc;
  }
}

// log_softmax over V=32000 + argmax (lowest-index ties), write all_outputs row
__global__ __launch_bounds__(256)
void k_logsoftmax(const float* __restrict__ logits, float* __restrict__ outp,
                  int* __restrict__ word_argmax) {
  int b = blockIdx.x, tid = threadIdx.x;
  const float* lb = logits + (size_t)b * NV;
  __shared__ float sv[256];
  __shared__ int   si[256];
  float mx = -3.4e38f; int mi = 0;
  for (int v = tid; v < NV; v += 256) {
    float x = lb[v];
    if (x > mx) { mx = x; mi = v; }
  }
  sv[tid] = mx; si[tid] = mi; __syncthreads();
  for (int off = 128; off > 0; off >>= 1) {
    if (tid < off) {
      float a = sv[tid], c = sv[tid + off];
      if (c > a || (c == a && si[tid + off] < si[tid])) { sv[tid] = c; si[tid] = si[tid + off]; }
    }
    __syncthreads();
  }
  float gmax = sv[0];
  if (tid == 0) word_argmax[b] = si[0];
  __syncthreads();
  float ps = 0.f;
  for (int v = tid; v < NV; v += 256) ps += __expf(lb[v] - gmax);
  sv[tid] = ps; __syncthreads();
  for (int off = 128; off > 0; off >>= 1) {
    if (tid < off) sv[tid] += sv[tid + off];
    __syncthreads();
  }
  float lse = gmax + __logf(sv[0]);
  for (int v = tid; v < NV; v += 256)
    outp[(size_t)b * (NT * NV) + v] = lb[v] - lse;    // all_outputs[b][t][v]
}

// greedy bookkeeping: word_preds / reached / unk_mask / last_out
__global__ void k_book(const float* __restrict__ uw, const int* __restrict__ unk_argmax,
                       const int* __restrict__ word_argmax,
                       const int* __restrict__ UNK_word_ids,
                       int* __restrict__ last_out, int* __restrict__ reached,
                       float* __restrict__ umask, float* __restrict__ d_pred) {
  int b = threadIdx.x;
  if (b >= NB) return;
  int isu = (uw[b * 17] < 0.5f) ? 1 : 0;
  int ua = unk_argmax[b];
  int wp = isu ? UNK_word_ids[b * NU + ua] : word_argmax[b];
  d_pred[(size_t)b * NT] = (float)wp;                 // all_preds[b][t]
  int r = reached[b];
  reached[b] = r + ((wp == 2) ? 1 : 0) * (1 - r);     // EOS_IDX == 2
  if (isu) umask[b * NU + ua] -= 1.f;
  last_out[b] = wp;
}

__global__ void k_emit_lengths(const int* __restrict__ lengths, float* __restrict__ d_len) {
  int b = threadIdx.x;
  if (b < NB) d_len[b] = (float)lengths[b];
}

// ---------------------------------------------------------------------------
extern "C" void kernel_launch(void* const* d_in, const int* in_sizes, int n_in,
                              void* d_out, int out_size, void* d_ws, size_t ws_size,
                              hipStream_t stream) {
  (void)in_sizes; (void)n_in; (void)out_size; (void)ws_size;
  const float* semantics    = (const float*)d_in[0];
  const float* styles       = (const float*)d_in[1];
  const float* UNK_embeds   = (const float*)d_in[2];
  const int*   UNK_lengths  = (const int*)  d_in[3];
  const int*   UNK_word_ids = (const int*)  d_in[4];
  const float* emb_table    = (const float*)d_in[5];
  const float* W_ih         = (const float*)d_in[6];
  const float* W_hh         = (const float*)d_in[7];
  const float* b_lstm       = (const float*)d_in[8];
  const float* W_att        = (const float*)d_in[9];
  const float* w_nounk      = (const float*)d_in[10];
  const float* W_out        = (const float*)d_in[11];
  const float* b_out        = (const float*)d_in[12];

  float* out = (float*)d_out;
  char*  ws  = (char*)d_ws;
  _Float16* wcat   = (_Float16*)(ws + OFF_WCAT);
  _Float16* woutt  = (_Float16*)(ws + OFF_WOUTT);
  _Float16* wattt  = (_Float16*)(ws + OFF_WATTT);
  _Float16* xcat   = (_Float16*)(ws + OFF_XCAT);
  float*    z      = (float*)   (ws + OFF_Z);
  float*    h      = (float*)   (ws + OFF_H);
  float*    c      = (float*)   (ws + OFF_C);
  _Float16* h16    = (_Float16*)(ws + OFF_H16);
  float*    q      = (float*)   (ws + OFF_Q);
  float*    logits = (float*)   (ws + OFF_LOGITS);
  float*    scores = (float*)   (ws + OFF_SCORES);
  float*    uw     = (float*)   (ws + OFF_UW);
  _Float16* lunk16 = (_Float16*)(ws + OFF_LUNK16);
  float*    umask  = (float*)   (ws + OFF_UMASK);
  int*      reach  = (int*)     (ws + OFF_REACH);
  int*      lens   = (int*)     (ws + OFF_LEN);
  int*      lastw  = (int*)     (ws + OFF_LAST);
  int*      warg   = (int*)     (ws + OFF_WARG);
  int*      uarg   = (int*)     (ws + OFF_UARG);

  // one-time f16 weight prep (re-done every launch: deterministic)
  k_build_wcat<<<(N4H * KCAT) / 256, 256, 0, stream>>>(W_ih, W_hh, wcat);
  k_transpose_f16<<<((size_t)NV * NH + 255) / 256, 256, 0, stream>>>(W_out, woutt, NH, NV);
  k_transpose_f16<<<((size_t)NE * NH + 255) / 256, 256, 0, stream>>>(W_att, wattt, NH, NE);
  k_init<<<256, 256, 0, stream>>>(h, c, h16, lunk16, umask, reach, lens, lastw);

  for (int t = 0; t < NT; ++t) {
    k_build_x<<<NB, 256, 0, stream>>>(emb_table, semantics, styles, lunk16, h16,
                                      lastw, reach, lens, xcat);
    // z = [x|h] @ Wcat^T + b_lstm   (K=3072, N=4096)
    k_gemm_wmma<<<dim3(N4H / 32, NB / 16), 32, 0, stream>>>(xcat, wcat, b_lstm, z, KCAT, N4H);
    k_lstm<<<(NB * NH) / 256, 256, 0, stream>>>(z, h, c, h16);
    // q = h @ W_att   (K=1024, N=512)
    k_gemm_wmma<<<dim3(NE / 32, NB / 16), 32, 0, stream>>>(h16, wattt, nullptr, q, NH, NE);
    k_scores<<<dim3(NU + 1, NB), 32, 0, stream>>>(q, h, UNK_embeds, w_nounk, scores);
    k_unk_softmax<<<NB, 32, 0, stream>>>(scores, UNK_lengths, umask, UNK_embeds,
                                         uw, lunk16, uarg,
                                         out + OUT_UW_BASE + (size_t)t * 17);
    // logits = h @ W_out + b_out   (K=1024, N=32000)
    k_gemm_wmma<<<dim3(NV / 32, NB / 16), 32, 0, stream>>>(h16, woutt, b_out, logits, NH, NV);
    k_logsoftmax<<<NB, 256, 0, stream>>>(logits, out + (size_t)t * NV, warg);
    k_book<<<1, 64, 0, stream>>>(uw, uarg, warg, UNK_word_ids, lastw, reach, umask,
                                 out + OUT_PRED_BASE + t);
  }
  k_emit_lengths<<<1, 64, 0, stream>>>(lens, out + OUT_LEN_BASE);
}
